// VanillaRNN_41721312313407
// MI455X (gfx1250) — matrix operand order repaired and checked by reading, not compiled, not run on hardware
//
#include <hip/hip_runtime.h>
#include <hip/hip_bf16.h>

// VanillaRNN on gfx1250 (MI455X), wave32 + WMMA bf16 path.
// B=1024, T=128, H=1024, C=10, D_IN=1.
//
// 64 workgroups x 16 batch rows (recurrence independent per row -> no grid sync).
// Per step, each WG computes [16 x 1024] = [16 x 1024] @ [1024 x 1024] with
// v_wmma_f32_16x16x32_bf16: h-tile in LDS (half-fragment A layout, 32 KB bf16,
// bank-conflict-free ds_load_b128), W_hh streamed from L2 in pre-packed
// B-fragment layout (coalesced global_load_b128), two-stage software-pipelined
// B prefetch with distinct register buffers (load->use distance = one full
// 4-wmma group, and across the barrier/tanh epilogue at step boundaries).

#define Bsz 1024
#define Tt  128
#define Hh  1024
#define Cc  10

typedef __attribute__((ext_vector_type(16))) __bf16 v16bf;
typedef __attribute__((ext_vector_type(8)))  __bf16 v8bf;
typedef __attribute__((ext_vector_type(8)))  float  v8f;

__device__ __forceinline__ float fast_tanh(float v) {
#if __has_builtin(__builtin_amdgcn_tanhf)
    return __builtin_amdgcn_tanhf(v);
#elif __has_builtin(__builtin_amdgcn_tanh_f32)
    return __builtin_amdgcn_tanh_f32(v);
#else
    return tanhf(v);
#endif
}

// ---------------------------------------------------------------------------
// Pack W_hh [H][H] (f32, row-major, rows = K, cols = N) into bf16 WMMA
// B-fragments in d_ws. Element (K, N) goes to flat bf16 index:
//   ((nt*32 + kc)*32 + lane)*16 + j
// with nt = N>>4, kc = K>>5, lane = ((K>>3)&1)*16 | (N&15), j = ((K&16)?8:0) | (K&7).
// (B 32x16 layout, ISA 7.12.2: lanes 0-15 hold K=0-7,16-23, lanes 16-31 K=8-15,24-31,
//  element j of lane group g -> K = (j>>3)*16 + g*8 + (j&7).)
// ---------------------------------------------------------------------------
__global__ void VanillaRNN_pack_whh(const float* __restrict__ whh,
                                    __bf16* __restrict__ pB) {
    int idx = blockIdx.x * blockDim.x + threadIdx.x;   // 0 .. H*H-1
    int K = idx >> 10;
    int N = idx & (Hh - 1);
    float v = whh[idx];
    int nt   = N >> 4;
    int kc   = K >> 5;
    int lane = (((K >> 3) & 1) << 4) | (N & 15);
    int j    = ((K & 16) ? 8 : 0) | (K & 7);
    pB[((((nt << 5) + kc) << 5) + lane) * 16 + j] = (__bf16)v;
}

// h-tile LDS layout (A half-fragments, bank-conflict-free):
//   element (kc, lane, j)  ->  kc*512 + (j>>3)*256 + lane*8 + (j&7)
// Each ds_load_b128 reads 16 B at lane*16 within a 512 B half-fragment:
// 16 lanes span all 64 LDS banks -> 2-phase service (optimal).
__device__ __forceinline__ int hA_idx(int kc, int lane, int j) {
    return kc * 512 + ((j >> 3) << 8) + lane * 8 + (j & 7);
}

// ---------------------------------------------------------------------------
// Main recurrent kernel: one WG = 16 batch rows, 512 threads = 16 waves.
// Wave w owns N-tiles {4w .. 4w+3}  (4x A-fragment reuse per LDS read pair).
// ---------------------------------------------------------------------------
__global__ void __launch_bounds__(512)
VanillaRNN_step_kernel(const float* __restrict__ x,
                       const float* __restrict__ whx,
                       const __bf16* __restrict__ pB,
                       const float* __restrict__ wph,
                       const float* __restrict__ bh,
                       const float* __restrict__ bp,
                       float* __restrict__ out) {
    __shared__ __align__(32) __bf16 hA[32 * 32 * 16];  // 32 KB
    __shared__ float xs[16][Tt];                       // 8 KB

    const int tid  = threadIdx.x;
    const int lane = tid & 31;
    const int wave = tid >> 5;
    const int row0 = blockIdx.x * 16;

    // h0 = 0
    for (int i = tid; i < 32 * 32 * 16; i += 512) hA[i] = (__bf16)0.0f;
    // stage this WG's 16 rows of x
    for (int i = tid; i < 16 * Tt; i += 512) {
        int m = i >> 7, t = i & (Tt - 1);
        xs[m][t] = x[(row0 + m) * Tt + t];
    }
    __syncthreads();

    // Per-lane constants. D-fragment: element e of v8f -> M = e + hi*8, N = n0 + cN.
    const int cN = lane & 15;
    const int hi = lane >> 4;
    float whxv[4], bhv[4];
#pragma unroll
    for (int i = 0; i < 4; ++i) {
        int N = (wave * 4 + i) * 16 + cN;
        whxv[i] = whx[N];
        bhv[i]  = bh[N];
    }
    // B-fragment base: nt stride = 32*32*16 = 16384 elems, kc stride = 512 elems
    const __bf16* pBw = pB + (size_t)(wave * 4) * 16384 + lane * 16;

    // Software-pipeline prologue: b0 <- kc=0 fragments. Because pB is constant
    // over time steps, the wrap-around reload at the end of each step's K-loop
    // re-fills b0 with kc=0, so this prologue runs exactly once.
    v16bf b0[4], b1[4];
#pragma unroll
    for (int i = 0; i < 4; ++i)
        b0[i] = *(const v16bf*)(pBw + (size_t)i * 16384);

    for (int t = 0; t < Tt; ++t) {
        v8f acc[4];
#pragma unroll
        for (int i = 0; i < 4; ++i) acc[i] = (v8f){};

        for (int kc = 0; kc < 32; kc += 2) {
            // prefetch kc+1 into the alternate buffer (distance: 4 wmma + ds)
#pragma unroll
            for (int i = 0; i < 4; ++i)
                b1[i] = *(const v16bf*)(pBw + (kc + 1) * 512 + (size_t)i * 16384);

            {   // A fragment kc: two bank-conflict-free half reads
                v8bf alo = *(const v8bf*)&hA[kc * 512 + lane * 8];
                v8bf ahi = *(const v8bf*)&hA[kc * 512 + 256 + lane * 8];
                v16bf a = __builtin_shufflevector(alo, ahi,
                            0, 1, 2, 3, 4, 5, 6, 7, 8, 9, 10, 11, 12, 13, 14, 15);
#pragma unroll
                for (int i = 0; i < 4; ++i)
                    acc[i] = __builtin_amdgcn_wmma_f32_16x16x32_bf16(
                        false, a, false, b0[i], (short)0, acc[i], false, false);
            }

            // prefetch kc+2 (wraps to kc=0 on the last iteration -> in-bounds,
            // and doubles as next step's prologue)
            const int kn = (kc + 2) & 31;
#pragma unroll
            for (int i = 0; i < 4; ++i)
                b0[i] = *(const v16bf*)(pBw + kn * 512 + (size_t)i * 16384);

            {   // A fragment kc+1
                v8bf alo = *(const v8bf*)&hA[(kc + 1) * 512 + lane * 8];
                v8bf ahi = *(const v8bf*)&hA[(kc + 1) * 512 + 256 + lane * 8];
                v16bf a = __builtin_shufflevector(alo, ahi,
                            0, 1, 2, 3, 4, 5, 6, 7, 8, 9, 10, 11, 12, 13, 14, 15);
#pragma unroll
                for (int i = 0; i < 4; ++i)
                    acc[i] = __builtin_amdgcn_wmma_f32_16x16x32_bf16(
                        false, a, false, b1[i], (short)0, acc[i], false, false);
            }
        }
        __syncthreads();   // all waves done reading old h

        // epilogue: h_new = tanh(acc + x_t*W_hx[N] + b_h[N]); scatter into
        // A half-fragment layout for next step (each wave owns disjoint kc's).
        float xv[8];
#pragma unroll
        for (int e = 0; e < 8; ++e) xv[e] = xs[hi * 8 + e][t];

#pragma unroll
        for (int i = 0; i < 4; ++i) {
            const int N   = (wave * 4 + i) * 16 + cN;   // = K index for next A
            const int kc  = N >> 5;
            const int k32 = N & 31;
            const int j   = ((k32 & 16) ? 8 : 0) | (k32 & 7);
            const int lg  = ((k32 & 15) >= 8) ? 16 : 0;
#pragma unroll
            for (int e = 0; e < 8; ++e) {
                float hv = fast_tanh(acc[i][e] + xv[e] * whxv[i] + bhv[i]);
                hA[hA_idx(kc, lg + (hi * 8 + e), j)] = (__bf16)hv;
            }
        }
        __syncthreads();   // new h visible to all waves
    }

    // Output projection: out[row0+m][c] = h[m][:] @ W_ph[:,c] + b_p[c]
    // (16x10 per WG; read h back out of the half-fragment layout.)
    if (tid < 16 * Cc) {
        const int m = tid / Cc, c = tid % Cc;
        float acc = bp[c];
        for (int K = 0; K < Hh; ++K) {
            const int k32 = K & 31;
            const int j   = ((k32 & 16) ? 8 : 0) | (k32 & 7);
            const int lg  = ((k32 & 15) >= 8) ? 16 : 0;
            float hv = (float)hA[hA_idx(K >> 5, lg + m, j)];
            acc += hv * wph[K * Cc + c];
        }
        out[(row0 + m) * Cc + c] = acc;
    }
}

extern "C" void kernel_launch(void* const* d_in, const int* in_sizes, int n_in,
                              void* d_out, int out_size, void* d_ws, size_t ws_size,
                              hipStream_t stream) {
    (void)in_sizes; (void)n_in; (void)out_size; (void)ws_size;
    const float* x   = (const float*)d_in[0];   // [B, T]
    const float* whx = (const float*)d_in[1];   // [1, H]
    const float* whh = (const float*)d_in[2];   // [H, H]
    const float* wph = (const float*)d_in[3];   // [H, C]
    const float* bh  = (const float*)d_in[4];   // [1, H]
    const float* bp  = (const float*)d_in[5];   // [1, C]
    __bf16* pB = (__bf16*)d_ws;                 // 2 MB packed W_hh (bf16 B-fragments)

    // one-time (per launch) repack of W_hh into WMMA B-fragment order
    VanillaRNN_pack_whh<<<dim3((Hh * Hh) / 256), dim3(256), 0, stream>>>(whh, pB);
    // persistent recurrent kernel: 64 WGs x 512 threads (16 waves)
    VanillaRNN_step_kernel<<<dim3(Bsz / 16), dim3(512), 0, stream>>>(
        x, whx, pB, wph, bh, bp, (float*)d_out);
}
